// ST_TR_Block_369367187599
// MI455X (gfx1250) — matrix operand order repaired and checked by reading, not compiled, not run on hardware
//
#include <hip/hip_runtime.h>
#include <hip/hip_bf16.h>
#include <math.h>

// ---------------------------------------------------------------------------
// Shapes (fixed by setup_inputs): B=8, T=64, E=128, C=256, H=2C=512, M=4C=1024
// ---------------------------------------------------------------------------
#define Bc 8
#define Tc 64
#define Ec 128
#define Cc 256
#define Hc 512
#define Mc 1024
#define NROWS (Bc * Tc * Ec)   // 65536

typedef __bf16 v16bf __attribute__((ext_vector_type(16)));
typedef float  v8f   __attribute__((ext_vector_type(8)));
typedef unsigned short u16;
typedef unsigned int   u32;
typedef u32   u32x4 __attribute__((ext_vector_type(4)));
typedef u32   u32x2 __attribute__((ext_vector_type(2)));
typedef float f32x4 __attribute__((ext_vector_type(4)));

static __device__ __forceinline__ u16 f2bf(float f) {
    u32 u = __float_as_uint(f);
    u32 r = u + 0x7FFFu + ((u >> 16) & 1u);   // round-to-nearest-even
    return (u16)(r >> 16);
}
static __device__ __forceinline__ float gelu_exact(float x) {
    return 0.5f * x * (1.0f + erff(x * 0.70710678118654752f));
}

union Frag { v16bf v; u32 u[8]; u32x4 q[2]; };

// ---------------------------------------------------------------------------
// Weight packing: fp32 -> bf16, transposed so Bt[n][k] (B^T row-major).
// ---------------------------------------------------------------------------
__global__ void pack_transpose_k(u16* __restrict__ dst, const float* __restrict__ src,
                                 int R, int Ccols) {
    int idx = blockIdx.x * 256 + threadIdx.x;
    if (idx >= R * Ccols) return;
    int c = idx / R;           // output row (= source column)
    int r = idx % R;           // output col (= source row)
    dst[idx] = f2bf(src[(size_t)r * Ccols + c]);
}
__global__ void pack_flat_k(u16* __restrict__ dst, const float* __restrict__ src, int n) {
    int idx = blockIdx.x * 256 + threadIdx.x;
    if (idx < n) dst[idx] = f2bf(src[idx]);
}
// conv1_w [H][C][3] -> conv1T[h][dt*C + c]  (im2col K order: dt-major, c contiguous)
__global__ void pack_conv1_k(u16* __restrict__ dst, const float* __restrict__ src) {
    int idx = blockIdx.x * 256 + threadIdx.x;
    if (idx >= Hc * 3 * Cc) return;
    int h   = idx / (3 * Cc);
    int rem = idx % (3 * Cc);
    int dt  = rem / Cc;
    int c   = rem % Cc;
    dst[idx] = f2bf(src[(size_t)h * (Cc * 3) + c * 3 + dt]);
}

// ---------------------------------------------------------------------------
// LayerNorm: one wave per row of 256, lane holds 8 elements (2x float4).
// ---------------------------------------------------------------------------
__global__ __launch_bounds__(256) void ln_k(const float* __restrict__ in,
                                            const float* __restrict__ w,
                                            const float* __restrict__ b,
                                            float* __restrict__ outF,
                                            u16* __restrict__ outBF) {
    int row  = blockIdx.x * 8 + (threadIdx.x >> 5);
    int lane = threadIdx.x & 31;
    const f32x4* p4 = (const f32x4*)(in + (size_t)row * Cc);
    f32x4 va = p4[lane], vb = p4[lane + 32];
    float s = va[0]+va[1]+va[2]+va[3]+vb[0]+vb[1]+vb[2]+vb[3];
#pragma unroll
    for (int m = 16; m > 0; m >>= 1) s += __shfl_xor(s, m, 32);
    float mean = s * (1.f / Cc);
    float q = 0.f;
#pragma unroll
    for (int i = 0; i < 4; ++i) { float d = va[i]-mean; q += d*d; d = vb[i]-mean; q += d*d; }
#pragma unroll
    for (int m = 16; m > 0; m >>= 1) q += __shfl_xor(q, m, 32);
    float rstd = rsqrtf(q * (1.f / Cc) + 1e-5f);
    f32x4 w0 = ((const f32x4*)w)[lane], w1 = ((const f32x4*)w)[lane + 32];
    f32x4 b0 = ((const f32x4*)b)[lane], b1 = ((const f32x4*)b)[lane + 32];
    f32x4 o0, o1;
#pragma unroll
    for (int i = 0; i < 4; ++i) {
        o0[i] = (va[i] - mean) * rstd * w0[i] + b0[i];
        o1[i] = (vb[i] - mean) * rstd * w1[i] + b1[i];
    }
    if (outF) {
        f32x4* of = (f32x4*)(outF + (size_t)row * Cc);
        of[lane] = o0; of[lane + 32] = o1;
    }
    if (outBF) {
        u32x2* ob = (u32x2*)(outBF + (size_t)row * Cc);
        u32x2 pk0, pk1;
        pk0[0] = (u32)f2bf(o0[0]) | ((u32)f2bf(o0[1]) << 16);
        pk0[1] = (u32)f2bf(o0[2]) | ((u32)f2bf(o0[3]) << 16);
        pk1[0] = (u32)f2bf(o1[0]) | ((u32)f2bf(o1[1]) << 16);
        pk1[1] = (u32)f2bf(o1[2]) | ((u32)f2bf(o1[3]) << 16);
        ob[lane] = pk0; ob[lane + 32] = pk1;
    }
}

// ---------------------------------------------------------------------------
// Spatial fuse: o = LN(p, on); x1 = x + xs + gamma_s*o -> d_out; xt = LN(x1,tn)
// ---------------------------------------------------------------------------
__global__ __launch_bounds__(256) void sfuse_k(const float* __restrict__ p,
                                               const float* __restrict__ on_w,
                                               const float* __restrict__ on_b,
                                               const float* __restrict__ x,
                                               const float* __restrict__ xs,
                                               const float* __restrict__ gamma_s,
                                               const float* __restrict__ tn_w,
                                               const float* __restrict__ tn_b,
                                               float* __restrict__ xout,
                                               u16* __restrict__ xtb) {
    int row  = blockIdx.x * 8 + (threadIdx.x >> 5);
    int lane = threadIdx.x & 31;
    size_t vbase = (size_t)row * 64;            // in float4 units
    const f32x4* p4 = (const f32x4*)p;
    f32x4 va = p4[vbase + lane], vb = p4[vbase + lane + 32];
    float s = va[0]+va[1]+va[2]+va[3]+vb[0]+vb[1]+vb[2]+vb[3];
#pragma unroll
    for (int m = 16; m > 0; m >>= 1) s += __shfl_xor(s, m, 32);
    float mean = s * (1.f / Cc);
    float q = 0.f;
#pragma unroll
    for (int i = 0; i < 4; ++i) { float d = va[i]-mean; q += d*d; d = vb[i]-mean; q += d*d; }
#pragma unroll
    for (int m = 16; m > 0; m >>= 1) q += __shfl_xor(q, m, 32);
    float rstd = rsqrtf(q * (1.f / Cc) + 1e-5f);
    float gs = gamma_s[0];
    f32x4 w0 = ((const f32x4*)on_w)[lane], w1 = ((const f32x4*)on_w)[lane + 32];
    f32x4 c0 = ((const f32x4*)on_b)[lane], c1 = ((const f32x4*)on_b)[lane + 32];
    f32x4 xa = ((const f32x4*)x)[vbase + lane],  xb = ((const f32x4*)x)[vbase + lane + 32];
    f32x4 sa = ((const f32x4*)xs)[vbase + lane], sb = ((const f32x4*)xs)[vbase + lane + 32];
    f32x4 x1a, x1b;
#pragma unroll
    for (int i = 0; i < 4; ++i) {
        x1a[i] = xa[i] + sa[i] + gs * ((va[i] - mean) * rstd * w0[i] + c0[i]);
        x1b[i] = xb[i] + sb[i] + gs * ((vb[i] - mean) * rstd * w1[i] + c1[i]);
    }
    float s2 = x1a[0]+x1a[1]+x1a[2]+x1a[3]+x1b[0]+x1b[1]+x1b[2]+x1b[3];
#pragma unroll
    for (int m = 16; m > 0; m >>= 1) s2 += __shfl_xor(s2, m, 32);
    float mean2 = s2 * (1.f / Cc);
    float q2 = 0.f;
#pragma unroll
    for (int i = 0; i < 4; ++i) { float d = x1a[i]-mean2; q2 += d*d; d = x1b[i]-mean2; q2 += d*d; }
#pragma unroll
    for (int m = 16; m > 0; m >>= 1) q2 += __shfl_xor(q2, m, 32);
    float rstd2 = rsqrtf(q2 * (1.f / Cc) + 1e-5f);
    f32x4 t0 = ((const f32x4*)tn_w)[lane], t1 = ((const f32x4*)tn_w)[lane + 32];
    f32x4 u0 = ((const f32x4*)tn_b)[lane], u1 = ((const f32x4*)tn_b)[lane + 32];
    ((f32x4*)xout)[vbase + lane]      = x1a;
    ((f32x4*)xout)[vbase + lane + 32] = x1b;
    u32x2 pk0, pk1;
    pk0[0] = (u32)f2bf((x1a[0]-mean2)*rstd2*t0[0]+u0[0]) |
             ((u32)f2bf((x1a[1]-mean2)*rstd2*t0[1]+u0[1]) << 16);
    pk0[1] = (u32)f2bf((x1a[2]-mean2)*rstd2*t0[2]+u0[2]) |
             ((u32)f2bf((x1a[3]-mean2)*rstd2*t0[3]+u0[3]) << 16);
    pk1[0] = (u32)f2bf((x1b[0]-mean2)*rstd2*t1[0]+u1[0]) |
             ((u32)f2bf((x1b[1]-mean2)*rstd2*t1[1]+u1[1]) << 16);
    pk1[1] = (u32)f2bf((x1b[2]-mean2)*rstd2*t1[2]+u1[2]) |
             ((u32)f2bf((x1b[3]-mean2)*rstd2*t1[3]+u1[3]) << 16);
    u32x2* ob = (u32x2*)(xtb + (size_t)row * Cc);
    ob[lane] = pk0; ob[lane + 32] = pk1;
}

// ---------------------------------------------------------------------------
// Generic WMMA bf16 GEMM: out[M,N] = A[M,K] * Bt[N,K]^T. 128x128 block tile,
// 8 waves (4x2), each wave 2x4 tiles of 16x16, K-step 32, double-buffered LDS,
// b128 LDS traffic, global_prefetch 2 steps ahead.
// ---------------------------------------------------------------------------
enum { AL_PLAIN = 0, AL_IM2COL = 1 };
enum { EPI_BF16 = 0, EPI_BIAS_F32 = 1, EPI_BN_GELU_BF16 = 2,
       EPI_BN_RESID = 3, EPI_BIAS_GELU_BF16 = 4, EPI_BIAS_RESID = 5 };

struct GemmP {
    const u16* A;      // plain: [M,K] bf16; im2col: xt [B,T,E,C] bf16
    const u16* Bt;     // [N,K] bf16 (transposed weights)
    float* outF;
    u16*   outBF;
    const float* p0;   // bias / bn weight
    const float* p1;   // bn bias
    const float* p2;   // bn mean
    const float* p3;   // bn var
    const float* scal; // gamma_t
    float* resid;      // d_out for residual epilogues
    int M, N, K;
};

template <int AL, int EPI>
__global__ __launch_bounds__(256) void gemm_k(GemmP P) {
    __shared__ u32 lA[2][128][20] __attribute__((aligned(16)));
    __shared__ u32 lB[2][128][20] __attribute__((aligned(16)));
    const int tid  = threadIdx.x;
    const int lane = tid & 31;
    const int w    = tid >> 5;
    const int wm   = w >> 1;     // 0..3
    const int wn   = w & 1;      // 0..1
    const int half = lane >> 4;
    const int l16  = lane & 15;
    const int mbase = blockIdx.y * 128;
    const int nbase = blockIdx.x * 128;
    const int r0 = tid >> 2;          // 0..63 (loads rows r0 and r0+64)
    const int c4 = (tid & 3) << 2;    // dword col base {0,4,8,12}; elem base c4*2

    v8f acc[2][4];
#pragma unroll
    for (int i = 0; i < 2; ++i)
#pragma unroll
        for (int j = 0; j < 4; ++j) acc[i][j] = (v8f){0.f,0.f,0.f,0.f,0.f,0.f,0.f,0.f};

    auto fetchA = [&](int ks, u32x4& x0, u32x4& x1) {
        if (AL == AL_PLAIN) {
            x0 = *(const u32x4*)(P.A + (size_t)(mbase + r0)      * P.K + ks * 32 + c4 * 2);
            x1 = *(const u32x4*)(P.A + (size_t)(mbase + r0 + 64) * P.K + ks * 32 + c4 * 2);
        } else {
            int dt = ks >> 3, cb = (ks & 7) << 5;
#pragma unroll
            for (int hh = 0; hh < 2; ++hh) {
                int g  = mbase + r0 + hh * 64;      // row in (b,e,t) order
                int t  = g & 63;
                int e  = (g >> 6) & 127;
                int bb = g >> 13;
                int tp = t + dt - 1;
                u32x4 val = {0u, 0u, 0u, 0u};
                if (tp >= 0 && tp < Tc)
                    val = *(const u32x4*)(P.A +
                        ((((size_t)(bb * Tc + tp)) * Ec + e) << 8) + cb + c4 * 2);
                if (hh == 0) x0 = val; else x1 = val;
            }
        }
    };
    auto fetchB = [&](int ks, u32x4& x0, u32x4& x1) {
        x0 = *(const u32x4*)(P.Bt + (size_t)(nbase + r0)      * P.K + ks * 32 + c4 * 2);
        x1 = *(const u32x4*)(P.Bt + (size_t)(nbase + r0 + 64) * P.K + ks * 32 + c4 * 2);
    };

    const int nk = P.K >> 5;
    {   u32x4 a0, a1, b0, b1;
        fetchA(0, a0, a1); fetchB(0, b0, b1);
        *(u32x4*)&lA[0][r0][c4]      = a0;
        *(u32x4*)&lA[0][r0 + 64][c4] = a1;
        *(u32x4*)&lB[0][r0][c4]      = b0;
        *(u32x4*)&lB[0][r0 + 64][c4] = b1;
    }
    __syncthreads();

    int p = 0;
    for (int ks = 0; ks < nk; ++ks) {
        const bool hasnext = (ks + 1 < nk);
        u32x4 na0, na1, nb0, nb1;
        if (hasnext) { fetchA(ks + 1, na0, na1); fetchB(ks + 1, nb0, nb1); }
        if (ks + 2 < nk) {
            if (AL == AL_PLAIN)
                __builtin_prefetch(P.A + (size_t)(mbase + r0) * P.K + (ks + 2) * 32 + c4 * 2, 0, 0);
            __builtin_prefetch(P.Bt + (size_t)(nbase + r0) * P.K + (ks + 2) * 32 + c4 * 2, 0, 0);
        }

        Frag fa[2], fb[4];
#pragma unroll
        for (int mt = 0; mt < 2; ++mt) {
            int row = wm * 32 + mt * 16 + l16;
            fa[mt].q[0] = *(const u32x4*)&lA[p][row][half * 4];
            fa[mt].q[1] = *(const u32x4*)&lA[p][row][8 + half * 4];
        }
#pragma unroll
        for (int j = 0; j < 4; ++j) {
            int col = wn * 64 + j * 16 + l16;
            fb[j].q[0] = *(const u32x4*)&lB[p][col][half * 4];
            fb[j].q[1] = *(const u32x4*)&lB[p][col][8 + half * 4];
        }
#pragma unroll
        for (int mt = 0; mt < 2; ++mt)
#pragma unroll
            for (int j = 0; j < 4; ++j)
                acc[mt][j] = __builtin_amdgcn_wmma_f32_16x16x32_bf16(
                    false, fa[mt].v, false, fb[j].v, (short)0, acc[mt][j], false, false);

        if (hasnext) {
            *(u32x4*)&lA[p ^ 1][r0][c4]      = na0;
            *(u32x4*)&lA[p ^ 1][r0 + 64][c4] = na1;
            *(u32x4*)&lB[p ^ 1][r0][c4]      = nb0;
            *(u32x4*)&lB[p ^ 1][r0 + 64][c4] = nb1;
        }
        __syncthreads();
        p ^= 1;
    }

    // ------------------------------- epilogue -------------------------------
#pragma unroll
    for (int mt = 0; mt < 2; ++mt) {
#pragma unroll
        for (int j = 0; j < 4; ++j) {
            int c = nbase + wn * 64 + j * 16 + l16;
            float bw = 0.f, bb2 = 0.f, bm = 0.f, brv = 0.f;
            if (EPI == EPI_BIAS_F32 || EPI == EPI_BIAS_GELU_BF16 || EPI == EPI_BIAS_RESID)
                bw = P.p0[c];
            if (EPI == EPI_BN_GELU_BF16 || EPI == EPI_BN_RESID) {
                bw = P.p0[c]; bb2 = P.p1[c]; bm = P.p2[c];
                brv = rsqrtf(P.p3[c] + 1e-5f);
            }
#pragma unroll
            for (int r = 0; r < 8; ++r) {
                int g = mbase + wm * 32 + mt * 16 + half * 8 + r;
                float v = acc[mt][j][r];
                if (EPI == EPI_BF16) {
                    P.outBF[(size_t)g * P.N + c] = f2bf(v);
                } else if (EPI == EPI_BIAS_F32) {
                    P.outF[(size_t)g * P.N + c] = v + bw;
                } else if (EPI == EPI_BN_GELU_BF16) {
                    float t = (v - bm) * brv * bw + bb2;
                    P.outBF[(size_t)g * P.N + c] = f2bf(gelu_exact(t));
                } else if (EPI == EPI_BN_RESID) {
                    float t = (v - bm) * brv * bw + bb2;
                    int tt = g & 63, e = (g >> 6) & 127, bi = g >> 13;
                    size_t oi = ((((size_t)(bi * Tc + tt)) * Ec + e) << 8) + c;
                    float x1 = P.resid[oi];
                    P.resid[oi] = 2.f * x1 + P.scal[0] * t;
                } else if (EPI == EPI_BIAS_GELU_BF16) {
                    P.outBF[(size_t)g * P.N + c] = f2bf(gelu_exact(v + bw));
                } else { // EPI_BIAS_RESID
                    size_t oi = (size_t)g * Cc + c;
                    P.resid[oi] = P.resid[oi] + v + bw;
                }
            }
        }
    }
}

// ---------------------------------------------------------------------------
// Attention: one workgroup per (b,t,head). E=128, D=32, 8 heads.
// attn = (q@k^T * mask + 1) * 0.5 ; o = attn @ v. All matmuls via WMMA bf16.
// ---------------------------------------------------------------------------
__global__ __launch_bounds__(256) void attn_k(const u16* __restrict__ qkv,
                                              const int* __restrict__ adj,
                                              u16* __restrict__ obuf) {
    __shared__ u32 qS[128][20] __attribute__((aligned(16)));
    __shared__ u32 kS[128][20] __attribute__((aligned(16)));
    __shared__ u16 vT[32][136] __attribute__((aligned(16)));   // v transposed: [d][f]
    __shared__ u16 aS[128][136] __attribute__((aligned(16)));  // attn bf16: [e][f]

    const int gid = blockIdx.x;
    const int h   = gid & 7;
    const int bt  = gid >> 3;
    const size_t rowbase = (size_t)bt * Ec;
    const int tid  = threadIdx.x;
    const int lane = tid & 31;
    const int w    = tid >> 5;
    const int half = lane >> 4;
    const int l16  = lane & 15;
    const int r0 = tid >> 2;          // 0..63
    const int c4 = (tid & 3) << 2;    // dword col base

    // load q,k tiles (128x32 bf16, b128) and transpose v into LDS
#pragma unroll
    for (int hh = 0; hh < 2; ++hh) {
        int e = r0 + hh * 64;
        const u16* base = qkv + (rowbase + e) * 768 + h * 32;
        u32x4 qv = *(const u32x4*)(base + c4 * 2);
        u32x4 kv = *(const u32x4*)(base + 256 + c4 * 2);
        u32x4 vv = *(const u32x4*)(base + 512 + c4 * 2);
        *(u32x4*)&qS[e][c4] = qv;
        *(u32x4*)&kS[e][c4] = kv;
#pragma unroll
        for (int i = 0; i < 4; ++i) {
            u32 w2 = vv[i];
            vT[(c4 + i) * 2 + 0][e] = (u16)(w2 & 0xFFFFu);
            vT[(c4 + i) * 2 + 1][e] = (u16)(w2 >> 16);
        }
    }
    __syncthreads();

    // GEMM1: attn[E,E] = q[E,32] @ k^T ; 8x8 tiles, wave w -> (em=w>>1, en=w&1)
    const int em = w >> 1, en = w & 1;
    v8f a1[2][4];
#pragma unroll
    for (int i = 0; i < 2; ++i)
#pragma unroll
        for (int j = 0; j < 4; ++j) a1[i][j] = (v8f){0.f,0.f,0.f,0.f,0.f,0.f,0.f,0.f};
    {
        Frag fq[2], fk[4];
#pragma unroll
        for (int mt = 0; mt < 2; ++mt) {
            int row = em * 32 + mt * 16 + l16;
            fq[mt].q[0] = *(const u32x4*)&qS[row][half * 4];
            fq[mt].q[1] = *(const u32x4*)&qS[row][8 + half * 4];
        }
#pragma unroll
        for (int j = 0; j < 4; ++j) {
            int col = en * 64 + j * 16 + l16;
            fk[j].q[0] = *(const u32x4*)&kS[col][half * 4];
            fk[j].q[1] = *(const u32x4*)&kS[col][8 + half * 4];
        }
#pragma unroll
        for (int mt = 0; mt < 2; ++mt)
#pragma unroll
            for (int j = 0; j < 4; ++j)
                a1[mt][j] = __builtin_amdgcn_wmma_f32_16x16x32_bf16(
                    false, fq[mt].v, false, fk[j].v, (short)0, a1[mt][j], false, false);
    }
    // affine mask, quantize to bf16 into aS
#pragma unroll
    for (int mt = 0; mt < 2; ++mt) {
#pragma unroll
        for (int j = 0; j < 4; ++j) {
            int f = en * 64 + j * 16 + l16;
#pragma unroll
            for (int r = 0; r < 8; ++r) {
                int e = em * 32 + mt * 16 + half * 8 + r;
                float mv = (float)adj[e * Ec + f];
                aS[e][f] = f2bf((a1[mt][j][r] * mv + 1.f) * 0.5f);
            }
        }
    }
    __syncthreads();

    // GEMM2: o[E,32] = attn[E,128] @ v[128,32]; wave w owns e-tile w, 2 d-tiles
    v8f a2[2];
    a2[0] = (v8f){0.f,0.f,0.f,0.f,0.f,0.f,0.f,0.f};
    a2[1] = (v8f){0.f,0.f,0.f,0.f,0.f,0.f,0.f,0.f};
#pragma unroll
    for (int s = 0; s < 4; ++s) {
        Frag fa, fv[2];
        int row = w * 16 + l16;
        fa.q[0] = *(const u32x4*)&aS[row][s * 32 + 8 * half];
        fa.q[1] = *(const u32x4*)&aS[row][s * 32 + 16 + 8 * half];
#pragma unroll
        for (int nt = 0; nt < 2; ++nt) {
            int dd = nt * 16 + l16;
            fv[nt].q[0] = *(const u32x4*)&vT[dd][s * 32 + 8 * half];
            fv[nt].q[1] = *(const u32x4*)&vT[dd][s * 32 + 16 + 8 * half];
        }
        a2[0] = __builtin_amdgcn_wmma_f32_16x16x32_bf16(false, fa.v, false, fv[0].v,
                                                        (short)0, a2[0], false, false);
        a2[1] = __builtin_amdgcn_wmma_f32_16x16x32_bf16(false, fa.v, false, fv[1].v,
                                                        (short)0, a2[1], false, false);
    }
#pragma unroll
    for (int nt = 0; nt < 2; ++nt) {
        int c = h * 32 + nt * 16 + l16;
#pragma unroll
        for (int r = 0; r < 8; ++r) {
            int e = w * 16 + half * 8 + r;
            obuf[(rowbase + e) * Cc + c] = f2bf(a2[nt][r]);
        }
    }
}

// ---------------------------------------------------------------------------
// Host-side launch
// ---------------------------------------------------------------------------
extern "C" void kernel_launch(void* const* d_in, const int* in_sizes, int n_in,
                              void* d_out, int out_size, void* d_ws, size_t ws_size,
                              hipStream_t stream) {
    (void)in_sizes; (void)n_in; (void)out_size; (void)ws_size;
    const float* x       = (const float*)d_in[0];
    const int*   adj     = (const int*)  d_in[1];
    const float* n1_w    = (const float*)d_in[2];
    const float* n1_b    = (const float*)d_in[3];
    const float* q_w     = (const float*)d_in[4];
    const float* k_w     = (const float*)d_in[5];
    const float* v_w     = (const float*)d_in[6];
    const float* proj_w  = (const float*)d_in[7];
    const float* proj_b  = (const float*)d_in[8];
    const float* on_w    = (const float*)d_in[9];
    const float* on_b    = (const float*)d_in[10];
    const float* gamma_s = (const float*)d_in[11];
    const float* conv1_w = (const float*)d_in[12];
    const float* bn1_w   = (const float*)d_in[13];
    const float* bn1_b   = (const float*)d_in[14];
    const float* bn1_m   = (const float*)d_in[15];
    const float* bn1_v   = (const float*)d_in[16];
    const float* conv2_w = (const float*)d_in[17];
    const float* bn2_w   = (const float*)d_in[18];
    const float* bn2_b   = (const float*)d_in[19];
    const float* bn2_m   = (const float*)d_in[20];
    const float* bn2_v   = (const float*)d_in[21];
    const float* tn_w    = (const float*)d_in[22];
    const float* tn_b    = (const float*)d_in[23];
    const float* gamma_t = (const float*)d_in[24];
    const float* n3_w    = (const float*)d_in[25];
    const float* n3_b    = (const float*)d_in[26];
    const float* fc1_w   = (const float*)d_in[27];
    const float* fc1_b   = (const float*)d_in[28];
    const float* fc2_w   = (const float*)d_in[29];
    const float* fc2_b   = (const float*)d_in[30];
    float* out = (float*)d_out;

    const size_t MiB = 1024ull * 1024ull;
    char* ws = (char*)d_ws;
    u16*   wbf  = (u16*)(ws + 0);          // packed bf16 weights (~2.5 MiB)
    float* xs_f = (float*)(ws + 4  * MiB); // 64 MiB  (later reused as y1)
    u16*   xs_b = (u16*)  (ws + 68 * MiB); // 32 MiB  (later reused as xm)
    u16*   qkv  = (u16*)  (ws + 100* MiB); // 96 MiB  (later reused as xt / h)
    u16*   obuf = (u16*)  (ws + 196* MiB); // 32 MiB
    float* pbuf = (float*)(ws + 228* MiB); // 64 MiB
    u16*   xtb  = (u16*)  (ws + 100* MiB); // 32 MiB (aliases dead qkv)
    u16*   y1   = (u16*)  (ws + 4  * MiB); // 64 MiB (aliases dead xs_f)
    u16*   xm   = (u16*)  (ws + 68 * MiB); // 32 MiB (aliases dead xs_b)
    u16*   hbuf = (u16*)  (ws + 132* MiB); // 128 MiB (aliases dead qkv tail/o/p)

    // packed weight sub-offsets (elements)
    u16* qkvT  = wbf + 0;            // [768][256]
    u16* projT = wbf + 196608;       // [256][256]
    u16* c1T   = wbf + 262144;       // [512][768]
    u16* c2T   = wbf + 655360;       // [256][512]
    u16* fc1T  = wbf + 786432;       // [1024][256]
    u16* fc2T  = wbf + 1048576;      // [256][1024]

    // ---- pack weights (bf16, transposed) ----
    pack_transpose_k<<<(256*256+255)/256, 256, 0, stream>>>(qkvT,          q_w, 256, 256);
    pack_transpose_k<<<(256*256+255)/256, 256, 0, stream>>>(qkvT + 65536,  k_w, 256, 256);
    pack_transpose_k<<<(256*256+255)/256, 256, 0, stream>>>(qkvT + 131072, v_w, 256, 256);
    pack_transpose_k<<<(256*256+255)/256, 256, 0, stream>>>(projT, proj_w, 256, 256);
    pack_conv1_k    <<<(393216+255)/256,  256, 0, stream>>>(c1T, conv1_w);
    pack_flat_k     <<<(131072+255)/256,  256, 0, stream>>>(c2T, conv2_w, 131072);
    pack_transpose_k<<<(256*1024+255)/256,256, 0, stream>>>(fc1T, fc1_w, 256, 1024);
    pack_transpose_k<<<(1024*256+255)/256,256, 0, stream>>>(fc2T, fc2_w, 1024, 256);

    // ---- LN1 ----
    ln_k<<<NROWS/8, 256, 0, stream>>>(x, n1_w, n1_b, xs_f, xs_b);

    // ---- QKV GEMM: [65536,256] x [256,768] -> qkv bf16 ----
    {   GemmP P{}; P.A = xs_b; P.Bt = qkvT; P.outBF = qkv;
        P.M = NROWS; P.N = 768; P.K = 256;
        gemm_k<AL_PLAIN, EPI_BF16><<<dim3(6, NROWS/128), 256, 0, stream>>>(P); }

    // ---- attention ----
    attn_k<<<Bc*Tc*8, 256, 0, stream>>>(qkv, adj, obuf);

    // ---- proj GEMM (+bias) -> pbuf f32 ----
    {   GemmP P{}; P.A = obuf; P.Bt = projT; P.outF = pbuf; P.p0 = proj_b;
        P.M = NROWS; P.N = 256; P.K = 256;
        gemm_k<AL_PLAIN, EPI_BIAS_F32><<<dim3(2, NROWS/128), 256, 0, stream>>>(P); }

    // ---- spatial fuse: LN(p)+residual -> d_out ; LN(tn) -> xtb bf16 ----
    sfuse_k<<<NROWS/8, 256, 0, stream>>>(pbuf, on_w, on_b, x, xs_f, gamma_s,
                                         tn_w, tn_b, out, xtb);

    // ---- conv1 as im2col GEMM: K=768, N=512, BN1+GELU -> y1 bf16 ----
    {   GemmP P{}; P.A = xtb; P.Bt = c1T; P.outBF = y1;
        P.p0 = bn1_w; P.p1 = bn1_b; P.p2 = bn1_m; P.p3 = bn1_v;
        P.M = NROWS; P.N = 512; P.K = 768;
        gemm_k<AL_IM2COL, EPI_BN_GELU_BF16><<<dim3(4, NROWS/128), 256, 0, stream>>>(P); }

    // ---- conv2 1x1 GEMM: K=512, N=256, BN2 + temporal residual into d_out ----
    {   GemmP P{}; P.A = y1; P.Bt = c2T; P.resid = out; P.scal = gamma_t;
        P.p0 = bn2_w; P.p1 = bn2_b; P.p2 = bn2_m; P.p3 = bn2_v;
        P.M = NROWS; P.N = 256; P.K = 512;
        gemm_k<AL_PLAIN, EPI_BN_RESID><<<dim3(2, NROWS/128), 256, 0, stream>>>(P); }

    // ---- LN3 -> xm bf16 ----
    ln_k<<<NROWS/8, 256, 0, stream>>>(out, n3_w, n3_b, nullptr, xm);

    // ---- fc1 GEMM (+bias, GELU) -> h bf16 ----
    {   GemmP P{}; P.A = xm; P.Bt = fc1T; P.outBF = hbuf; P.p0 = fc1_b;
        P.M = NROWS; P.N = 1024; P.K = 256;
        gemm_k<AL_PLAIN, EPI_BIAS_GELU_BF16><<<dim3(8, NROWS/128), 256, 0, stream>>>(P); }

    // ---- fc2 GEMM (+bias) + residual add into d_out ----
    {   GemmP P{}; P.A = hbuf; P.Bt = fc2T; P.resid = out; P.p0 = fc2_b;
        P.M = NROWS; P.N = 256; P.K = 1024;
        gemm_k<AL_PLAIN, EPI_BIAS_RESID><<<dim3(2, NROWS/128), 256, 0, stream>>>(P); }
}
